// ParallelMambaModel_87179246174645
// MI455X (gfx1250) — compile-verified, matrix operand order checked by reading, not run on hardware
//
#include <hip/hip_runtime.h>

// ---------------- model constants ----------------
static constexpr int BB  = 16;          // batch
static constexpr int LL  = 8192;        // sequence length
static constexpr int DM  = 64;          // d_model
static constexpr int NS  = 16;          // d_state
static constexpr int DC  = 4;           // d_conv
static constexpr int DI  = 128;         // d_inner
static constexpr int NC  = 10;          // num classes
static constexpr int NCAT = DI + 2 * NS;        // 160 fused output features (dt|B|C)
static constexpr long long BL = (long long)BB * LL;   // 131072 rows

// ---------------- workspace layout (per branch) ----------------
static constexpr size_t SZ_XC16 = (size_t)BL * DI * sizeof(_Float16);  // 32 MB
static constexpr size_t SZ_BC   = (size_t)BL * NS * sizeof(float);     //  8 MB
static constexpr size_t OFF_XC  = 0;
static constexpr size_t OFF_ZS  = SZ_XC16;
static constexpr size_t OFF_DT  = 2 * SZ_XC16;
static constexpr size_t OFF_B   = 3 * SZ_XC16;
static constexpr size_t OFF_C   = 3 * SZ_XC16 + SZ_BC;
static constexpr size_t BR_STRIDE = 3 * SZ_XC16 + 2 * SZ_BC;           // 112 MB / branch
// weights region after the two branch data blocks
static constexpr size_t SZ_EFFW = 256 * 2 * sizeof(float);
static constexpr size_t SZ_EFFB = 256 * sizeof(float);
static constexpr size_t SZ_WCAT = (size_t)NCAT * DI * sizeof(_Float16);
static constexpr size_t WBR_STRIDE = SZ_EFFW + SZ_EFFB + SZ_WCAT;
static constexpr size_t OFF_W    = 2 * BR_STRIDE;
static constexpr size_t OFF_POOL = OFF_W + 2 * WBR_STRIDE;             // 2*16*128 f32

// ---------------- WMMA types ----------------
typedef __attribute__((ext_vector_type(16))) _Float16 v16h;
typedef __attribute__((ext_vector_type(8)))  float    v8f;
union FragH { v16h v; unsigned u[8]; };

// branchless softplus via hardware transcendentals (v_exp_f32 + v_log_f32)
__device__ __forceinline__ float softplus_fast(float x) {
  return fmaxf(x, 0.f) + __logf(1.f + __expf(-fabsf(x)));
}

// =====================================================================
// Kernel A: fold weights.
//   effw[c][k] = in_proj_w[c,:] @ proj_w[:,k]   (collapses input proj)
//   effb[c]    = in_proj_w[c,:] @ proj_b
//   W_cat[n][c]: n<128 -> dt_proj_w @ x_proj_w[0:4]; 128..143 -> B rows;
//                144..159 -> C rows.  Stored f16 for WMMA B-operand.
// =====================================================================
__global__ __launch_bounds__(256) void prep_kernel(
    const float* __restrict__ inw, const float* __restrict__ pw,
    const float* __restrict__ pb, int nch,
    const float* __restrict__ xpw, const float* __restrict__ dtw,
    float* __restrict__ effw, float* __restrict__ effb,
    _Float16* __restrict__ wcat)
{
  int t = threadIdx.x;
  if (t < 256) {
    float bsum = 0.f, w0 = 0.f, w1 = 0.f;
    for (int m = 0; m < DM; ++m) {
      float iv = inw[t * DM + m];
      bsum += iv * pb[m];
      w0 += iv * pw[m * nch + 0];
      if (nch > 1) w1 += iv * pw[m * nch + 1];
    }
    effb[t] = bsum;
    effw[t * 2 + 0] = w0;
    effw[t * 2 + 1] = (nch > 1) ? w1 : 0.f;
  }
  for (int idx = t; idx < NCAT * DI; idx += 256) {
    int n = idx / DI, c = idx % DI;
    float v;
    if (n < DI) {
      v = 0.f;
      for (int j = 0; j < 4; ++j) v += dtw[n * 4 + j] * xpw[j * DI + c];
    } else if (n < DI + NS) {
      v = xpw[(4 + (n - DI)) * DI + c];
    } else {
      v = xpw[(4 + NS + (n - DI - NS)) * DI + c];
    }
    wcat[idx] = (_Float16)v;
  }
}

// =====================================================================
// Kernel B: fused front end, one wave per 16-row tile.
//   per row l: xz from effective weights, causal depthwise conv (4 taps),
//   SiLU -> xc (LDS + global f16); silu(z) -> global f16.
//   Then WMMA GEMM: [16 x 128] x [128 x 160] with v_wmma_f32_16x16x32_f16,
//   softplus(+dt_bias) on the dt columns, B/C columns to f32 buffers.
// =====================================================================
__global__ __launch_bounds__(32) void front_kernel(
    const float* __restrict__ data, int nch,
    const float* __restrict__ effw, const float* __restrict__ effb,
    const float* __restrict__ convw, const float* __restrict__ convb,
    const float* __restrict__ dtb, const _Float16* __restrict__ wcat,
    _Float16* __restrict__ xc16, _Float16* __restrict__ zs16,
    _Float16* __restrict__ dt16, float* __restrict__ Bm, float* __restrict__ Cm)
{
  __shared__ _Float16 xcT[16][DI];
  const int lane = threadIdx.x;
  const long long row0 = (long long)blockIdx.x * 16;

  // ---- fused projection + conv + SiLU ----
  for (int idx = lane; idx < 16 * DI; idx += 32) {
    int r = idx >> 7, d = idx & 127;
    long long gl = row0 + r;
    int b = (int)(gl / LL), l = (int)(gl % LL);
    const float* dp = data + (long long)b * nch * LL;
    float w0 = effw[d * 2], w1 = effw[d * 2 + 1], eb = effb[d];
    float conv = convb[d];
#pragma unroll
    for (int j = 0; j < DC; ++j) {
      int lt = l - (DC - 1) + j;
      if (lt >= 0) {
        float xv = w0 * dp[lt] + eb;
        if (nch > 1) xv += w1 * dp[LL + lt];
        conv += convw[d * DC + j] * xv;
      }
    }
    float xc = conv / (1.f + __expf(-conv));          // SiLU
    xcT[r][d] = (_Float16)xc;
    xc16[gl * DI + d] = (_Float16)xc;
    float zw0 = effw[(d + DI) * 2], zw1 = effw[(d + DI) * 2 + 1];
    float z = zw0 * dp[l] + effb[d + DI];
    if (nch > 1) z += zw1 * dp[LL + l];
    zs16[gl * DI + d] = (_Float16)(z / (1.f + __expf(-z)));
  }
  __syncthreads();

  // ---- WMMA: D = xc_tile (16x128) * W_cat^T (128x160) ----
  const int laneLo = lane & 15, hi = lane >> 4;
  for (int nt = 0; nt < NCAT / 16; ++nt) {
    v8f acc = {};
    const _Float16* wr = wcat + (size_t)(nt * 16 + laneLo) * DI;
#pragma unroll
    for (int kt = 0; kt < DI / 32; ++kt) {
      FragH a, bf;
#pragma unroll
      for (int i = 0; i < 8; ++i) {
        // 16-bit A layout: lanes 0-15 hold K{0..7,16..23}, lanes 16-31 K{8..15,24..31}
        int kp = ((i < 4) ? 2 * i : 16 + 2 * (i - 4)) + (hi ? 8 : 0) + kt * 32;
        a.u[i]  = *reinterpret_cast<const unsigned*>(&xcT[laneLo][kp]);
        bf.u[i] = *reinterpret_cast<const unsigned*>(wr + kp);
      }
      acc = __builtin_amdgcn_wmma_f32_16x16x32_f16(
          false, a.v, false, bf.v, (short)0, acc, false, false);
    }
    int ccol = nt * 16 + laneLo;
#pragma unroll
    for (int r = 0; r < 8; ++r) {
      long long gl = row0 + r + (hi ? 8 : 0);
      float v = acc[r];
      if (ccol < DI) {                                // dt: + bias, softplus
        dt16[gl * DI + ccol] = (_Float16)softplus_fast(v + dtb[ccol]);
      } else if (ccol < DI + NS) {
        Bm[gl * NS + (ccol - DI)] = v;
      } else {
        Cm[gl * NS + (ccol - DI - NS)] = v;
      }
    }
  }
}

// =====================================================================
// Kernel C: selective scan. 1 thread per (branch, b, d); h[16] + A row in
// registers; B/C staged per 64-step chunk in LDS; accumulates the gated,
// D-skipped output mean directly (pooling commutes with out_proj).
// =====================================================================
static constexpr int CH = 64;

__global__ __launch_bounds__(128) void scan_kernel(
    const char* __restrict__ ws,
    const float* __restrict__ Alog_t, const float* __restrict__ Alog_f,
    const float* __restrict__ D_t,    const float* __restrict__ D_f)
{
  const int b = blockIdx.x, br = blockIdx.y, d = threadIdx.x;
  const char* base = ws + (size_t)br * BR_STRIDE;
  const _Float16* xc16 = (const _Float16*)(base + OFF_XC);
  const _Float16* zs16 = (const _Float16*)(base + OFF_ZS);
  const _Float16* dt16 = (const _Float16*)(base + OFF_DT);
  const float*    Bm   = (const float*)(base + OFF_B);
  const float*    Cm   = (const float*)(base + OFF_C);
  float* pooled = (float*)(const_cast<char*>(ws) + OFF_POOL);

  const float* Alog = br ? Alog_f : Alog_t;
  const float  Dd   = (br ? D_f : D_t)[d];

  float Ar[NS];
#pragma unroll
  for (int n = 0; n < NS; ++n) Ar[n] = -__expf(Alog[d * NS + n]);
  float h[NS];
#pragma unroll
  for (int n = 0; n < NS; ++n) h[n] = 0.f;
  float acc = 0.f;

  __shared__ float sB[CH][NS];
  __shared__ float sC[CH][NS];

  for (int l0 = 0; l0 < LL; l0 += CH) {
    const long long rbase = (long long)b * LL + l0;
    const float* gB = Bm + rbase * NS;
    const float* gC = Cm + rbase * NS;
    for (int i = d; i < CH * NS; i += 128) {
      sB[i / NS][i % NS] = gB[i];
      sC[i / NS][i % NS] = gC[i];
    }
    __syncthreads();
    if (d == 0 && l0 + CH < LL) {                 // hide next chunk's latency
      __builtin_prefetch(gB + CH * NS, 0, 1);     // -> global_prefetch_b8
      __builtin_prefetch(gC + CH * NS, 0, 1);
    }
    for (int i = 0; i < CH; ++i) {
      long long gl = rbase + i;
      float dtv = (float)dt16[gl * DI + d];
      float xv  = (float)xc16[gl * DI + d];
      float zv  = (float)zs16[gl * DI + d];
      float dtx = dtv * xv;
      float y = 0.f;
#pragma unroll
      for (int n = 0; n < NS; ++n) {              // 16-way ILP across states
        float dA = __expf(dtv * Ar[n]);
        h[n] = fmaf(dA, h[n], dtx * sB[i][n]);
        y = fmaf(h[n], sC[i][n], y);
      }
      acc = fmaf(y + xv * Dd, zv, acc);           // (+xc*D) * silu(z), pooled
    }
    __syncthreads();
  }
  pooled[((size_t)br * BB + b) * DI + d] = acc * (1.f / LL);
}

// =====================================================================
// Kernel D: finalize. pooled -> out_proj (tiny, post-pool) -> combined;
// combined -> fc logits. Writes d_out = [logits 16x10][combined 16x128].
// =====================================================================
__global__ __launch_bounds__(256) void final_kernel(
    const char* __restrict__ ws,
    const float* __restrict__ ow_t, const float* __restrict__ ow_f,
    const float* __restrict__ fcw, const float* __restrict__ fcb,
    float* __restrict__ out)
{
  __shared__ float comb[BB][2 * DM];
  const float* pooled = (const float*)(ws + OFF_POOL);
  int t = threadIdx.x;
  for (int idx = t; idx < BB * 2 * DM; idx += 256) {
    int b = idx / (2 * DM), j = idx % (2 * DM);
    int br = j / DM, m = j % DM;
    const float* W  = br ? ow_f : ow_t;
    const float* pv = pooled + ((size_t)br * BB + b) * DI;
    float v = 0.f;
    for (int dd = 0; dd < DI; ++dd) v += W[m * DI + dd] * pv[dd];
    comb[b][j] = v;
    out[BB * NC + idx] = v;                       // combined output
  }
  __syncthreads();
  for (int idx = t; idx < BB * NC; idx += 256) {
    int b = idx / NC, c = idx % NC;
    float v = fcb[c];
    for (int j = 0; j < 2 * DM; ++j) v += fcw[c * 2 * DM + j] * comb[b][j];
    out[idx] = v;                                 // logits
  }
}

// =====================================================================
extern "C" void kernel_launch(void* const* d_in, const int* in_sizes, int n_in,
                              void* d_out, int out_size, void* d_ws, size_t ws_size,
                              hipStream_t stream) {
  (void)in_sizes; (void)n_in; (void)out_size; (void)ws_size;
  // input order (nested dicts flattened in insertion order)
  const float* time_data   = (const float*)d_in[0];
  const float* freq_data   = (const float*)d_in[1];
  const float* time_proj_w = (const float*)d_in[2];
  const float* time_proj_b = (const float*)d_in[3];
  const float* freq_proj_w = (const float*)d_in[4];
  const float* freq_proj_b = (const float*)d_in[5];
  // tparams: 6..14, fparams: 15..23
  const float* t_inw  = (const float*)d_in[6];
  const float* t_cw   = (const float*)d_in[7];
  const float* t_cb   = (const float*)d_in[8];
  const float* t_xpw  = (const float*)d_in[9];
  const float* t_dtw  = (const float*)d_in[10];
  const float* t_dtb  = (const float*)d_in[11];
  const float* t_Alog = (const float*)d_in[12];
  const float* t_D    = (const float*)d_in[13];
  const float* t_ow   = (const float*)d_in[14];
  const float* f_inw  = (const float*)d_in[15];
  const float* f_cw   = (const float*)d_in[16];
  const float* f_cb   = (const float*)d_in[17];
  const float* f_xpw  = (const float*)d_in[18];
  const float* f_dtw  = (const float*)d_in[19];
  const float* f_dtb  = (const float*)d_in[20];
  const float* f_Alog = (const float*)d_in[21];
  const float* f_D    = (const float*)d_in[22];
  const float* f_ow   = (const float*)d_in[23];
  const float* fc_w   = (const float*)d_in[24];
  const float* fc_b   = (const float*)d_in[25];

  char* ws = (char*)d_ws;
  // per-branch workspace pointers
  auto brp = [&](int br, size_t off) { return ws + (size_t)br * BR_STRIDE + off; };
  float*    effw0 = (float*)(ws + OFF_W);
  float*    effb0 = (float*)(ws + OFF_W + SZ_EFFW);
  _Float16* wcat0 = (_Float16*)(ws + OFF_W + SZ_EFFW + SZ_EFFB);
  float*    effw1 = (float*)(ws + OFF_W + WBR_STRIDE);
  float*    effb1 = (float*)(ws + OFF_W + WBR_STRIDE + SZ_EFFW);
  _Float16* wcat1 = (_Float16*)(ws + OFF_W + WBR_STRIDE + SZ_EFFW + SZ_EFFB);

  // A: fold weights (tiny)
  prep_kernel<<<1, 256, 0, stream>>>(t_inw, time_proj_w, time_proj_b, 1,
                                     t_xpw, t_dtw, effw0, effb0, wcat0);
  prep_kernel<<<1, 256, 0, stream>>>(f_inw, freq_proj_w, freq_proj_b, 2,
                                     f_xpw, f_dtw, effw1, effb1, wcat1);

  // B: fused front + WMMA GEMM, one wave per 16-row tile
  const int tiles = (int)(BL / 16);
  front_kernel<<<tiles, 32, 0, stream>>>(
      time_data, 1, effw0, effb0, t_cw, t_cb, t_dtb, wcat0,
      (_Float16*)brp(0, OFF_XC), (_Float16*)brp(0, OFF_ZS),
      (_Float16*)brp(0, OFF_DT), (float*)brp(0, OFF_B), (float*)brp(0, OFF_C));
  front_kernel<<<tiles, 32, 0, stream>>>(
      freq_data, 2, effw1, effb1, f_cw, f_cb, f_dtb, wcat1,
      (_Float16*)brp(1, OFF_XC), (_Float16*)brp(1, OFF_ZS),
      (_Float16*)brp(1, OFF_DT), (float*)brp(1, OFF_B), (float*)brp(1, OFF_C));

  // C: selective scan (both branches in one grid)
  scan_kernel<<<dim3(BB, 2), 128, 0, stream>>>(ws, t_Alog, f_Alog, t_D, f_D);

  // D: finalize
  final_kernel<<<1, 256, 0, stream>>>(ws, t_ow, f_ow, fc_w, fc_b, (float*)d_out);
}
